// FlowMatchingPolicyHead_50105088475482
// MI455X (gfx1250) — matrix operand order repaired and checked by reading, not compiled, not run on hardware
//
#include <hip/hip_runtime.h>

// ---------------------------------------------------------------------------
// FlowMatchingPolicyHead fused kernel for gfx1250 (MI455X, wave32, WMMA).
//
// Round 3: fix async builtin pointer types. The probe showed
// __builtin_amdgcn_global_load_async_to_lds_b128 exists and expects
// (v4i addrspace(1)*, v4i addrspace(3)*, imm, imm).
//
// Structure: workgroup-shared, double-buffered B staging in LDS via CDNA5
// async loads (GLOBAL_LOAD_ASYNC_TO_LDS_B128 / ASYNCcnt). 4 waves per WG
// split 32 rows x 256 cols (each wave: 16 rows x 128 cols, acc = 64 VGPRs),
// all waves share one 16KB B tile per k-step -> 4x less L2 traffic and no
// per-WMMA s_wait_loadcnt 0 stalls.
// ---------------------------------------------------------------------------

typedef _Float16 v16h __attribute__((ext_vector_type(16)));
typedef _Float16 h8   __attribute__((ext_vector_type(8)));
typedef float    v8f  __attribute__((ext_vector_type(8)));
typedef int      v4i  __attribute__((ext_vector_type(4)));

#define Bn 131072
#define An 23

// d_ws layout (units: _Float16)
static constexpr size_t OFF_WIN  = 0;                       // 256 x 352 (W_in^T, padded)
static constexpr size_t OFF_W1   = OFF_WIN + 256*352;       // 3 x 256 x 256 (W1^T)
static constexpr size_t OFF_W2   = OFF_W1  + 3*256*256;     // 3 x 256 x 256 (W2^T)
static constexpr size_t OFF_WV1  = OFF_W2  + 3*256*256;     // 256 x 256     (Wv1^T)
static constexpr size_t OFF_WOUT = OFF_WV1 + 256*256;       // 32  x 256     (W_out^T, rows 23..31 zero)
static constexpr size_t WS_HALFS = OFF_WOUT + 32*256;       // 557056 halfs

// ---------------- CDNA5 async-to-LDS capability probe ----------------------
#if __has_builtin(__builtin_amdgcn_global_load_async_to_lds_b128)
#define HAVE_ASYNC_LDS 1
#else
#define HAVE_ASYNC_LDS 0
#warning "CDNA5 probe: __builtin_amdgcn_global_load_async_to_lds_b128 NOT available; using sync staging fallback"
#endif
#if __has_builtin(__builtin_amdgcn_s_wait_asynccnt)
#define HAVE_WAIT_ASYNC 1
#else
#define HAVE_WAIT_ASYNC 0
#warning "CDNA5 probe: __builtin_amdgcn_s_wait_asynccnt NOT available; using inline asm"
#endif

typedef __attribute__((address_space(1))) v4i* gv4p;   // global 128-bit chunk ptr
typedef __attribute__((address_space(3))) v4i* lv4p;   // LDS    128-bit chunk ptr

__device__ __forceinline__ void async_cp16(const _Float16* g, _Float16* l) {
#if HAVE_ASYNC_LDS
    __builtin_amdgcn_global_load_async_to_lds_b128((gv4p)g, (lv4p)l, 0, 0);
#else
    *(h8*)l = *(const h8*)g;       // sync fallback: global_load_b128 + ds_store_b128
#endif
}

template<int N>
__device__ __forceinline__ void wait_async() {
#if HAVE_ASYNC_LDS
#if HAVE_WAIT_ASYNC
    __builtin_amdgcn_s_wait_asynccnt((short)N);
#else
    asm volatile("s_wait_asynccnt %0" :: "n"(N) : "memory");
#endif
#endif
}

// ---------------------------------------------------------------------------
// Prep: transpose + fp32->f16 convert all weight matrices into d_ws; zero loss.
// ---------------------------------------------------------------------------
__global__ void prep_kernel(const float* __restrict__ W_in,
                            const float* __restrict__ W1,
                            const float* __restrict__ W2,
                            const float* __restrict__ Wv1,
                            const float* __restrict__ W_out,
                            _Float16* __restrict__ ws,
                            float* __restrict__ out)
{
    size_t idx = (size_t)blockIdx.x * blockDim.x + threadIdx.x;
    if (idx == 0) out[0] = 0.0f;
    if (idx < 256*352) {
        int n = (int)(idx / 352), k = (int)(idx % 352);
        float v = 0.0f;
        if (k < 23)       v = W_in[(size_t)k * 256 + n];
        else if (k >= 32) v = W_in[(size_t)(k - 9) * 256 + n];
        ws[OFF_WIN + idx] = (_Float16)v;
        return;
    }
    idx -= 256*352;
    if (idx < 3*65536) {
        int i = (int)(idx / 65536), r = (int)(idx % 65536);
        int n = r / 256, k = r % 256;
        ws[OFF_W1 + (size_t)i*65536 + (size_t)n*256 + k] =
            (_Float16)W1[(size_t)i*65536 + (size_t)k*256 + n];
        return;
    }
    idx -= 3*65536;
    if (idx < 3*65536) {
        int i = (int)(idx / 65536), r = (int)(idx % 65536);
        int n = r / 256, k = r % 256;
        ws[OFF_W2 + (size_t)i*65536 + (size_t)n*256 + k] =
            (_Float16)W2[(size_t)i*65536 + (size_t)k*256 + n];
        return;
    }
    idx -= 3*65536;
    if (idx < 65536) {
        int n = (int)(idx / 256), k = (int)(idx % 256);
        ws[OFF_WV1 + idx] = (_Float16)Wv1[(size_t)k*256 + n];
        return;
    }
    idx -= 65536;
    if (idx < 32*256) {
        int n = (int)(idx / 256), k = (int)(idx % 256);
        float v = (n < 23) ? W_out[(size_t)k*23 + n] : 0.0f;
        ws[OFF_WOUT + idx] = (_Float16)v;
        return;
    }
}

// ---------------------------------------------------------------------------
__device__ __forceinline__ v8f wmma16(v16h a, v16h b, v8f c) {
    return __builtin_amdgcn_wmma_f32_16x16x32_f16(false, a, false, b, (short)0, c, false, false);
}

__device__ __forceinline__ float silu(float v) {
    return v / (1.0f + __expf(-v));
}

// Stage one 256x32-half B tile (16KB) into LDS slot `sb`, 16B chunks.
// LDS layout: chunk(part 0..3, row 0..255) at halfs ((part<<8)+row)*8 so the
// per-lane ds_load_b128 fragment reads below are bank-conflict-free.
__device__ __forceinline__ void stage_tile(_Float16* sb, const _Float16* __restrict__ Bt,
                                           int ldk, int ks)
{
    const int tid = threadIdx.x;
#pragma unroll
    for (int i = 0; i < 8; ++i) {
        int c = i*128 + tid;
        int row = c >> 2, part = c & 3;
        async_cp16(Bt + (size_t)row*ldk + ks*32 + part*8,
                   sb + (size_t)(((part << 8) + row) * 8));
    }
}

// Workgroup-cooperative staged GEMM: each wave computes 16 rows x 128 cols
// (8 tiles), B tile shared by all 4 waves through double-buffered LDS.
template<int KS, class AL>
__device__ __forceinline__ void wgemm_staged(v8f acc[8], const _Float16* __restrict__ Bt,
                                             _Float16 (*stg)[8192],
                                             int ncb, int col, int lh, AL aload)
{
    const int ldk = KS * 32;
    stage_tile(stg[0], Bt, ldk, 0);
    stage_tile(stg[1], Bt, ldk, 1);
    for (int ks = 0; ks < KS; ++ks) {
        if (ks == KS - 1) wait_async<0>(); else wait_async<8>();
        __syncthreads();
        v16h a = aload(ks);
        const _Float16* sb = stg[ks & 1];
#pragma unroll
        for (int n = 0; n < 8; ++n) {
            int ncol = ncb + n*16 + col;
            h8 blo = *(const h8*)(sb + (((lh    ) << 8) + ncol) * 8);
            h8 bhi = *(const h8*)(sb + (((lh + 2) << 8) + ncol) * 8);
            v16h b;
#pragma unroll
            for (int e = 0; e < 8; ++e) { b[e] = blo[e]; b[e+8] = bhi[e]; }
            acc[n] = wmma16(a, b, acc[n]);
        }
        __syncthreads();
        if (ks + 2 < KS) stage_tile(stg[ks & 1], Bt, ldk, ks + 2);
    }
}

// Direct-global small GEMM (W_out), per-wave, no barriers.
template<int NT, int KS, class AL>
__device__ __forceinline__ void wgemm_direct(v8f acc[NT], const _Float16* __restrict__ Bt,
                                             int col, int lh, AL aload)
{
    const int ldk = KS * 32;
#pragma unroll
    for (int ks = 0; ks < KS; ++ks) {
        v16h a = aload(ks);
#pragma unroll
        for (int n = 0; n < NT; ++n) {
            const _Float16* bp = Bt + (size_t)(n*16 + col) * ldk + ks*32 + lh*8;
            h8 blo = *(const h8*)bp;
            h8 bhi = *(const h8*)(bp + 16);
            v16h b;
#pragma unroll
            for (int e = 0; e < 8; ++e) { b[e] = blo[e]; b[e+8] = bhi[e]; }
            acc[n] = wmma16(a, b, acc[n]);
        }
    }
}

__device__ __forceinline__ void zero8(v8f acc[8]) {
#pragma unroll
    for (int n = 0; n < 8; ++n)
#pragma unroll
        for (int e = 0; e < 8; ++e) acc[n][e] = 0.0f;
}

// ---------------------------------------------------------------------------
// Fused main kernel: 4 waves/block; wave (strip, ch) owns rows strip*16..+16
// and cols ch*128..+128. 32 rows/block -> grid 4096.
// ---------------------------------------------------------------------------
__global__ __launch_bounds__(128) void
flow_fused_kernel(const float* __restrict__ state,
                  const float* __restrict__ actions,
                  const float* __restrict__ tarr,
                  const float* __restrict__ x0,
                  const float* __restrict__ b_in,
                  const float* __restrict__ ln_g,
                  const float* __restrict__ ln_b,
                  const float* __restrict__ b1,
                  const float* __restrict__ b2,
                  const float* __restrict__ b_out,
                  const float* __restrict__ bv1,
                  const float* __restrict__ Wv2,
                  const float* __restrict__ bv2,
                  const _Float16* __restrict__ ws,
                  float* __restrict__ out)
{
    __shared__ _Float16 xbuf[2][16][256];   // residual activations (f16), 16KB
    __shared__ _Float16 hbuf[2][16][256];   // block intermediate   (f16), 16KB
    __shared__ _Float16 stg[2][8192];       // double-buffered B tile,    32KB

    const int lane  = threadIdx.x & 31;
    const int wave  = threadIdx.x >> 5;
    const int lh    = lane >> 4;            // lane half (K split)
    const int col   = lane & 15;            // C col within tile / A row
    const int strip = wave & 1;             // row strip (16 rows)
    const int ch    = wave >> 1;            // column half (128 cols)
    const int ncb   = ch * 128;
    const int rowbase = (blockIdx.x * 2 + strip) * 16;
    const int grow  = rowbase + col;

    _Float16* xr = &xbuf[strip][col][0];
    _Float16* hr = &hbuf[strip][col][0];
    const float tval = tarr[grow];

    v8f acc[8];
    zero8(acc);

    // ---------------- Layer 0: x = silu(concat(x_t, t_emb, state) @ W_in + b_in)
    wgemm_staged<11>(acc, ws + OFF_WIN, stg, ncb, col, lh, [&](int ks) -> v16h {
        v16h a;
        const int kb = ks*32 + lh*8;
        if (ks >= 3) {                                   // state, k in [96,352)
            const float* p = state + (size_t)grow*256 + (kb - 96);
#pragma unroll
            for (int e = 0; e < 8; ++e) {
                a[e]   = (_Float16)p[e];
                a[e+8] = (_Float16)p[e + 16];
            }
        } else if (ks == 0) {                            // x_t (k<23), zero pad
#pragma unroll
            for (int r = 0; r < 2; ++r) {
#pragma unroll
                for (int e = 0; e < 8; ++e) {
                    int k = kb + r*16 + e;
                    float v = 0.0f;
                    if (k < 23) {
                        float xa = x0[(size_t)grow*23 + k];
                        float aa = actions[(size_t)grow*23 + k];
                        v = (1.0f - tval) * xa + tval * aa;
                    }
                    a[r*8 + e] = (_Float16)v;
                }
            }
        } else {                                         // time embedding [32,96)
#pragma unroll
            for (int r = 0; r < 2; ++r) {
#pragma unroll
                for (int e = 0; e < 8; ++e) {
                    int k  = kb + r*16 + e;
                    int kp = k - 32;
                    float fr  = __expf(-0.28782313662425575f * (float)(kp & 31));
                    float arg = tval * fr;
                    float v = (kp < 32) ? __sinf(arg) : __cosf(arg);
                    a[r*8 + e] = (_Float16)v;
                }
            }
        }
        return a;
    });

#pragma unroll
    for (int n = 0; n < 8; ++n) {
        int cc = ncb + n*16 + col;
        float bb = b_in[cc];
#pragma unroll
        for (int j = 0; j < 8; ++j)
            xbuf[strip][lh*8 + j][cc] = (_Float16)silu(acc[n][j] + bb);
    }

    // ---------------- 3 residual blocks: x += silu(LN(x)@W1+b1)@W2 + b2
    for (int blk = 0; blk < 3; ++blk) {
        __syncthreads();                    // xbuf halves written by both ch-waves
        // row stats: lane pair (col, col+16) covers row `col` of this strip
        float s = 0.0f, s2 = 0.0f;
        const _Float16* rr = &xbuf[strip][col][lh * 128];
#pragma unroll
        for (int c = 0; c < 128; c += 8) {
            h8 v8 = *(const h8*)(rr + c);
#pragma unroll
            for (int e = 0; e < 8; ++e) {
                float v = (float)v8[e];
                s += v; s2 += v * v;
            }
        }
        s  += __shfl_xor(s,  16, 32);
        s2 += __shfl_xor(s2, 16, 32);
        float mu   = s * (1.0f / 256.0f);
        float var  = s2 * (1.0f / 256.0f) - mu * mu;
        float rstd = rsqrtf(var + 1e-5f);
        const float* gp = ln_g + blk*256;
        const float* bp = ln_b + blk*256;

        zero8(acc);
        wgemm_staged<8>(acc, ws + OFF_W1 + (size_t)blk*65536, stg, ncb, col, lh,
                        [&](int ks) -> v16h {
            v16h a;
            const int kb = ks*32 + lh*8;
#pragma unroll
            for (int r = 0; r < 2; ++r)
#pragma unroll
                for (int e = 0; e < 8; ++e) {
                    int k = kb + r*16 + e;
                    float v = ((float)xr[k] - mu) * rstd * gp[k] + bp[k];
                    a[r*8 + e] = (_Float16)v;
                }
            return a;
        });

#pragma unroll
        for (int n = 0; n < 8; ++n) {
            int cc = ncb + n*16 + col;
            float bb = b1[blk*256 + cc];
#pragma unroll
            for (int j = 0; j < 8; ++j)
                hbuf[strip][lh*8 + j][cc] = (_Float16)silu(acc[n][j] + bb);
        }

        zero8(acc);
        wgemm_staged<8>(acc, ws + OFF_W2 + (size_t)blk*65536, stg, ncb, col, lh,
                        [&](int ks) -> v16h {
            v16h a;
            const int kb = ks*32 + lh*8;
            h8 lo = *(const h8*)(hr + kb);
            h8 hi = *(const h8*)(hr + kb + 16);
            v16h r;
#pragma unroll
            for (int e = 0; e < 8; ++e) { r[e] = lo[e]; r[e+8] = hi[e]; }
            return r;
        });

#pragma unroll
        for (int n = 0; n < 8; ++n) {       // residual: only this wave touches these cells
            int cc = ncb + n*16 + col;
            float bb = b2[blk*256 + cc];
#pragma unroll
            for (int j = 0; j < 8; ++j) {
                int rrow = lh*8 + j;
                float v = acc[n][j] + bb + (float)xbuf[strip][rrow][cc];
                xbuf[strip][rrow][cc] = (_Float16)v;
            }
        }
    }

    __syncthreads();                        // full rows of xbuf visible to all

    // ---------------- W_out + flow loss (ch==0 waves, per strip)
    if (ch == 0) {
        v8f oacc[2];
#pragma unroll
        for (int n = 0; n < 2; ++n)
#pragma unroll
            for (int e = 0; e < 8; ++e) oacc[n][e] = 0.0f;

        wgemm_direct<2, 8>(oacc, ws + OFF_WOUT, col, lh, [&](int ks) -> v16h {
            v16h a;
            const int kb = ks*32 + lh*8;
            h8 lo = *(const h8*)(xr + kb);
            h8 hi = *(const h8*)(xr + kb + 16);
#pragma unroll
            for (int e = 0; e < 8; ++e) { a[e] = lo[e]; a[e+8] = hi[e]; }
            return a;
        });

        float lsum = 0.0f;
#pragma unroll
        for (int n = 0; n < 2; ++n) {
            int cc = n*16 + col;
            if (cc < 23) {
                float bo = b_out[cc];
#pragma unroll
                for (int j = 0; j < 8; ++j) {
                    int rrow = rowbase + lh*8 + j;
                    float tgt = actions[(size_t)rrow*23 + cc] - x0[(size_t)rrow*23 + cc];
                    float d = oacc[n][j] + bo - tgt;
                    lsum += d * d;
                }
            }
        }
        lsum += __shfl_xor(lsum, 1, 32);
        lsum += __shfl_xor(lsum, 2, 32);
        lsum += __shfl_xor(lsum, 4, 32);
        lsum += __shfl_xor(lsum, 8, 32);
        lsum += __shfl_xor(lsum, 16, 32);
        if (lane == 0)
            atomicAdd(out, lsum * (1.0f / ((float)Bn * (float)An)));
    }

    // ---------------- Value head: silu(state@Wv1+bv1) @ Wv2 + bv2 (all waves)
    zero8(acc);
    wgemm_staged<8>(acc, ws + OFF_WV1, stg, ncb, col, lh, [&](int ks) -> v16h {
        v16h a;
        const int kb = ks*32 + lh*8;
        const float* p = state + (size_t)grow*256 + kb;
#pragma unroll
        for (int e = 0; e < 8; ++e) {
            a[e]   = (_Float16)p[e];
            a[e+8] = (_Float16)p[e + 16];
        }
        return a;
    });

    float vs[8];
#pragma unroll
    for (int j = 0; j < 8; ++j) vs[j] = 0.0f;
#pragma unroll
    for (int n = 0; n < 8; ++n) {
        int cc = ncb + n*16 + col;
        float bb = bv1[cc];
        float w2 = Wv2[cc];
#pragma unroll
        for (int j = 0; j < 8; ++j)
            vs[j] += silu(acc[n][j] + bb) * w2;
    }
#pragma unroll
    for (int j = 0; j < 8; ++j) {
        vs[j] += __shfl_xor(vs[j], 1, 32);
        vs[j] += __shfl_xor(vs[j], 2, 32);
        vs[j] += __shfl_xor(vs[j], 4, 32);
        vs[j] += __shfl_xor(vs[j], 8, 32);
    }

    __syncthreads();                        // stage buffer free -> overlay reduction scratch
    float* vred = (float*)&stg[0][0];       // [strip][ch][row16] = 64 floats
#pragma unroll
    for (int j = 0; j < 8; ++j)
        if (col == j) vred[strip*32 + ch*16 + lh*8 + j] = vs[j];
    __syncthreads();
    if (ch == 0 && lane < 16)
        out[1 + rowbase + lane] = vred[strip*32 + lane] + vred[strip*32 + 16 + lane] + bv2[0];
}

// ---------------------------------------------------------------------------
extern "C" void kernel_launch(void* const* d_in, const int* in_sizes, int n_in,
                              void* d_out, int out_size, void* d_ws, size_t ws_size,
                              hipStream_t stream)
{
    const float* state   = (const float*)d_in[0];
    const float* actions = (const float*)d_in[1];
    const float* tarr    = (const float*)d_in[2];
    const float* x0      = (const float*)d_in[3];
    const float* W_in    = (const float*)d_in[4];
    const float* b_in    = (const float*)d_in[5];
    const float* ln_g    = (const float*)d_in[6];
    const float* ln_b    = (const float*)d_in[7];
    const float* W1      = (const float*)d_in[8];
    const float* b1      = (const float*)d_in[9];
    const float* W2      = (const float*)d_in[10];
    const float* b2      = (const float*)d_in[11];
    const float* W_out   = (const float*)d_in[12];
    const float* b_out   = (const float*)d_in[13];
    const float* Wv1     = (const float*)d_in[14];
    const float* bv1     = (const float*)d_in[15];
    const float* Wv2     = (const float*)d_in[16];
    const float* bv2     = (const float*)d_in[17];

    _Float16* ws = (_Float16*)d_ws;
    float* out = (float*)d_out;

    prep_kernel<<<((int)WS_HALFS + 255) / 256, 256, 0, stream>>>(
        W_in, W1, W2, Wv1, W_out, ws, out);

    flow_fused_kernel<<<Bn / 32, 128, 0, stream>>>(
        state, actions, tarr, x0, b_in, ln_g, ln_b, b1, b2, b_out,
        bv1, Wv2, bv2, (const _Float16*)ws, out);
}
